// SingleStreamBlock_79345225826636
// MI455X (gfx1250) — matrix operand order repaired and checked by reading, not compiled, not run on hardware
//
#include <hip/hip_runtime.h>
#include <hip/hip_bf16.h>
#include <cstdint>

// CDNA5 / gfx1250 single-stream transformer block.
// Heavy math through v_wmma_f32_16x16x32_bf16 (wave32, fp32 accum).
// GEMMs use double-buffered LDS staging fed by global_load_async_to_lds_b128
// (ASYNCcnt-tracked) -- the CDNA5 async producer/consumer path.

typedef __attribute__((ext_vector_type(16))) __bf16 v16bf;
typedef __attribute__((ext_vector_type(8)))  __bf16 v8bf;
typedef __attribute__((ext_vector_type(8)))  float  v8f;

#define B_   2
#define L_   2048
#define H_   2048
#define NH_  16
#define D_   128
#define MLP_ 8192
#define N1_  (3 * H_ + MLP_)   // 14336
#define K2_  (H_ + MLP_)       // 10240
#define M_   (B_ * L_)         // 4096

__device__ __forceinline__ v8f wmma_bf16f32(v16bf a, v16bf b, v8f c) {
  return __builtin_amdgcn_wmma_f32_16x16x32_bf16(
      /*neg_a=*/false, a, /*neg_b=*/false, b,
      /*c_mod=*/(short)0, c, /*reuse_a=*/false, /*reuse_b=*/false);
}

// 16-bit A-matrix fragment (16x32 MxK): lane row = lane%16,
// 16 contiguous K halves starting at (lane>>4)*16.
__device__ __forceinline__ v16bf make_afrag(const __bf16* p) {
  v8bf lo = *(const v8bf*)(p);
  v8bf hi = *(const v8bf*)(p + 8);
  v16bf f;
#pragma unroll
  for (int e = 0; e < 8; ++e) { f[e] = lo[e]; f[e + 8] = hi[e]; }
  return f;
}

// 16-bit B-matrix fragment (32x16 KxN): lane col = lane%16,
// 16 contiguous K halves starting at (lane>>4)*16. Loaded as 2x16B
// (LDS rows are 80B-strided: 16B aligned, not 32B).
__device__ __forceinline__ v16bf make_bfrag(const __bf16* p) {
  v8bf lo = *(const v8bf*)(p);
  v8bf hi = *(const v8bf*)(p + 8);
  v16bf f;
#pragma unroll
  for (int e = 0; e < 8; ++e) { f[e] = lo[e]; f[e + 8] = hi[e]; }
  return f;
}

// ---------------- fp32 -> bf16 convert ----------------
__global__ __launch_bounds__(256) void cvt_bf16(const float* __restrict__ s,
                                                __bf16* __restrict__ d, size_t n) {
  size_t i = (size_t)blockIdx.x * 256 + threadIdx.x;
  if (i < n) d[i] = (__bf16)s[i];
}

// ---------------- mod = silu(vec) @ mod_w^T + mod_b ----------------
__global__ __launch_bounds__(256) void mod_kernel(const float* __restrict__ vec,
                                                  const float* __restrict__ mw,
                                                  const float* __restrict__ mb,
                                                  float* __restrict__ modo) {
  int o = blockIdx.x * 256 + threadIdx.x;      // 0 .. 2*6144-1
  int b = o / (3 * H_), j = o % (3 * H_);
  const float* vr = vec + (size_t)b * H_;
  const float* wr = mw + (size_t)j * H_;
  float acc = 0.f;
  for (int k = 0; k < H_; ++k) {
    float vv = vr[k];
    acc += (vv / (1.f + __expf(-vv))) * wr[k];  // silu
  }
  modo[o] = acc + mb[j];
}

// ---------------- x_mod = (1+scale)*LN(x) + shift  -> bf16 ----------------
__global__ __launch_bounds__(256) void ln_mod(const float* __restrict__ x,
                                              const float* __restrict__ modv,
                                              __bf16* __restrict__ xm) {
  __shared__ float sh[16];
  int row = blockIdx.x;                 // b*L + l
  int b = row >> 11;
  int t = threadIdx.x, lane = t & 31, wid = t >> 5;
  const float* xr = x + (size_t)row * H_;
  float v[8], s = 0.f, s2 = 0.f;
#pragma unroll
  for (int u = 0; u < 8; ++u) {
    v[u] = xr[t + u * 256];
    s += v[u]; s2 += v[u] * v[u];
  }
#pragma unroll
  for (int off = 16; off >= 1; off >>= 1) {
    s += __shfl_xor(s, off, 32);
    s2 += __shfl_xor(s2, off, 32);
  }
  if (lane == 0) { sh[wid] = s; sh[8 + wid] = s2; }
  __syncthreads();
  if (t == 0) {
    float a = 0.f, c = 0.f;
    for (int i = 0; i < 8; ++i) { a += sh[i]; c += sh[8 + i]; }
    sh[0] = a; sh[8] = c;
  }
  __syncthreads();
  float mean = sh[0] * (1.f / H_);
  float var  = sh[8] * (1.f / H_) - mean * mean;
  float inv  = rsqrtf(var + 1e-6f);
  const float* shift = modv + (size_t)b * 3 * H_;
  const float* scl   = shift + H_;
#pragma unroll
  for (int u = 0; u < 8; ++u) {
    int c = t + u * 256;
    float xn = (v[u] - mean) * inv;
    xm[(size_t)row * H_ + c] = (__bf16)((1.f + scl[c]) * xn + shift[c]);
  }
}

// ---------------- WMMA GEMM: C = A(MxK) * Bw(NxK)^T + bias ----------------
// 256 thr = 8 waves (4 M x 2 N); wave tile 32x64; block tile 128x128.
// Double-buffered LDS tiles filled by global_load_async_to_lds_b128.
// LDS rows padded 32->40 bf16 (80B stride) for bank-conflict-free frag reads.
template <bool FUSE>
__global__ __launch_bounds__(256) void gemm_bf16(const __bf16* __restrict__ A,
                                                 const __bf16* __restrict__ Bw,
                                                 const float* __restrict__ bias,
                                                 float* __restrict__ Cout,
                                                 const float* __restrict__ Xres,
                                                 const float* __restrict__ modv,
                                                 int M, int N, int K) {
  __shared__ __align__(16) __bf16 sA[2][128 * 40];
  __shared__ __align__(16) __bf16 sB[2][128 * 40];
  const int tid = threadIdx.x;
  const int lane = tid & 31, wid = tid >> 5;
  const int wm = wid & 3, wn = wid >> 2;
  const int mblk = blockIdx.x * 128, nblk = blockIdx.y * 128;
  const int l15 = lane & 15, lh = lane >> 4;

  v8f acc[2][4];
#pragma unroll
  for (int i = 0; i < 2; ++i)
#pragma unroll
    for (int j = 0; j < 4; ++j) acc[i][j] = {};

  const int S = K >> 5;  // K-steps of 32

  // Stage loader: each lane moves 2x16B of A and 2x16B of B (4 async ops/wave).
  auto issue = [&](int stage, int buf) {
    const int ks = stage << 5;
#pragma unroll
    for (int u = 0; u < 2; ++u) {
      int c = tid + u * 256;            // 0..511 chunk id
      int row = c >> 2;                 // 128 rows, 4 chunks (of 8 bf16) per row
      int ko = (c & 3) << 3;
      unsigned la = (unsigned)(uintptr_t)(&sA[buf][row * 40 + ko]);
      unsigned long long ga =
          (unsigned long long)(uintptr_t)(A + (size_t)(mblk + row) * K + ks + ko);
      asm volatile("global_load_async_to_lds_b128 %0, %1, off"
                   :: "v"(la), "v"(ga) : "memory");
      unsigned lb = (unsigned)(uintptr_t)(&sB[buf][row * 40 + ko]);
      unsigned long long gb =
          (unsigned long long)(uintptr_t)(Bw + (size_t)(nblk + row) * K + ks + ko);
      asm volatile("global_load_async_to_lds_b128 %0, %1, off"
                   :: "v"(lb), "v"(gb) : "memory");
    }
  };

  issue(0, 0);
  for (int s = 0; s < S; ++s) {
    const int buf = s & 1;
    if (s + 1 < S) {
      issue(s + 1, buf ^ 1);
      // async loads complete in order: <=4 outstanding leaves only next stage
      asm volatile("s_wait_asynccnt 4" ::: "memory");
    } else {
      asm volatile("s_wait_asynccnt 0" ::: "memory");
    }
    __syncthreads();
    v16bf fa0 = make_afrag(&sA[buf][(wm * 32 + l15) * 40 + lh * 16]);
    v16bf fa1 = make_afrag(&sA[buf][(wm * 32 + 16 + l15) * 40 + lh * 16]);
#pragma unroll
    for (int t = 0; t < 4; ++t) {
      v16bf fb = make_bfrag(&sB[buf][(wn * 64 + t * 16 + l15) * 40 + lh * 16]);
      acc[0][t] = wmma_bf16f32(fa0, fb, acc[0][t]);
      acc[1][t] = wmma_bf16f32(fa1, fb, acc[1][t]);
    }
    __syncthreads();
  }

  const int m0 = mblk + wm * 32;
  const int bb = m0 >> 11;  // batch index, uniform within tile
#pragma unroll
  for (int t = 0; t < 4; ++t) {
    int col = nblk + wn * 64 + t * 16 + l15;
    float bv = bias[col];
    float g = 0.f;
    if (FUSE) g = modv[(size_t)bb * 3 * H_ + 2 * H_ + col];
#pragma unroll
    for (int r = 0; r < 8; ++r) {
      int row0 = m0 + r + lh * 8;
      int row1 = row0 + 16;
      if (!FUSE) {
        Cout[(size_t)row0 * N + col] = acc[0][t][r] + bv;
        Cout[(size_t)row1 * N + col] = acc[1][t][r] + bv;
      } else {
        Cout[(size_t)row0 * N + col] =
            Xres[(size_t)row0 * N + col] + g * (acc[0][t][r] + bv);
        Cout[(size_t)row1 * N + col] =
            Xres[(size_t)row1 * N + col] + g * (acc[1][t][r] + bv);
      }
    }
  }
}

// ---------------- qkv extract + rmsnorm + rope; V stored transposed ----------------
__global__ __launch_bounds__(256) void qkv_prep(const float* __restrict__ h,
                                                const float* __restrict__ pe,
                                                const float* __restrict__ qs,
                                                const float* __restrict__ ks,
                                                __bf16* __restrict__ Qh,
                                                __bf16* __restrict__ Kh,
                                                __bf16* __restrict__ Vt) {
  int row = blockIdx.x;                 // b*L + l
  int b = row >> 11, l = row & (L_ - 1);
  int t = threadIdx.x;
  int hd = t >> 4, j = t & 15;          // 16 heads x 16 lanes
  int d0 = j * 8;
  const float* hb = h + (size_t)row * N1_;
  float q[8], k[8], v[8];
#pragma unroll
  for (int u = 0; u < 8; ++u) {
    q[u] = hb[hd * D_ + d0 + u];
    k[u] = hb[H_ + hd * D_ + d0 + u];
    v[u] = hb[2 * H_ + hd * D_ + d0 + u];
  }
  float sq = 0.f, sk = 0.f;
#pragma unroll
  for (int u = 0; u < 8; ++u) { sq += q[u] * q[u]; sk += k[u] * k[u]; }
#pragma unroll
  for (int off = 8; off >= 1; off >>= 1) {   // reduce within 16-lane head group
    sq += __shfl_xor(sq, off, 32);
    sk += __shfl_xor(sk, off, 32);
  }
  float rq = rsqrtf(sq * (1.f / D_) + 1e-6f);
  float rk = rsqrtf(sk * (1.f / D_) + 1e-6f);
#pragma unroll
  for (int u = 0; u < 8; ++u) {
    q[u] *= rq * qs[d0 + u];
    k[u] *= rk * ks[d0 + u];
  }
  const float* peb = pe + (size_t)(b * L_ + l) * (D_ / 2) * 4;
  float qo[8], ko[8];
#pragma unroll
  for (int p = 0; p < 4; ++p) {
    int i = (d0 >> 1) + p;
    float p00 = peb[i * 4 + 0], p01 = peb[i * 4 + 1];
    float p10 = peb[i * 4 + 2], p11 = peb[i * 4 + 3];
    qo[2 * p]     = p00 * q[2 * p] + p01 * q[2 * p + 1];
    qo[2 * p + 1] = p10 * q[2 * p] + p11 * q[2 * p + 1];
    ko[2 * p]     = p00 * k[2 * p] + p01 * k[2 * p + 1];
    ko[2 * p + 1] = p10 * k[2 * p] + p11 * k[2 * p + 1];
  }
  size_t qb = (((size_t)(b * NH_ + hd)) * L_ + l) * D_ + d0;
#pragma unroll
  for (int u = 0; u < 8; ++u) { Qh[qb + u] = (__bf16)qo[u]; Kh[qb + u] = (__bf16)ko[u]; }
  size_t vb = (size_t)(b * NH_ + hd) * D_ * L_;
#pragma unroll
  for (int u = 0; u < 8; ++u) Vt[vb + (size_t)(d0 + u) * L_ + l] = (__bf16)v[u];
}

// ---------------- gelu(mlp) -> y[:, H..H+MLP) ----------------
__global__ __launch_bounds__(256) void gelu_kernel(const float* __restrict__ h,
                                                   __bf16* __restrict__ y) {
  size_t i = (size_t)blockIdx.x * 256 + threadIdx.x;  // over M_*MLP_
  size_t row = i >> 13;                               // / MLP_
  int c = (int)(i & (MLP_ - 1));
  float xv = h[row * N1_ + 3 * H_ + c];
  float x3 = xv * xv * xv;
  float g = 0.5f * xv * (1.f + tanhf(0.7978845608f * (xv + 0.044715f * x3)));
  y[row * K2_ + H_ + c] = (__bf16)g;
}

// ---------------- flash attention (WMMA) -> y[:, 0..H) ----------------
// 128 thr = 4 waves; wave owns 16 q rows; q-tile = 64 rows per block.
__global__ __launch_bounds__(128) void attn_kernel(const __bf16* __restrict__ Q,
                                                   const __bf16* __restrict__ Kh,
                                                   const __bf16* __restrict__ Vt,
                                                   __bf16* __restrict__ Y) {
  __shared__ __align__(16) __bf16 pbuf[4][16 * 32];   // per-wave P tile (16x32)
  const int lane = threadIdx.x & 31, wid = threadIdx.x >> 5;
  const int bh = blockIdx.x;            // b*NH + h
  const int b = bh >> 4, hd = bh & 15;
  const int q0 = blockIdx.y * 64 + wid * 16;
  const int l15 = lane & 15, lh = lane >> 4;
  const size_t kvbase = (size_t)bh * L_ * D_;

  v16bf qf[4];
  const size_t qrow = kvbase + (size_t)(q0 + l15) * D_;
#pragma unroll
  for (int s = 0; s < 4; ++s) qf[s] = make_afrag(Q + qrow + s * 32 + lh * 16);

  v8f o[8] = {};
  float mi[8], li[8];
#pragma unroll
  for (int r = 0; r < 8; ++r) { mi[r] = -1e30f; li[r] = 0.f; }
  const float scale = 0.08838834764831845f;  // 1/sqrt(128)

  for (int kc = 0; kc < L_; kc += 32) {
    v8f s0 = {}, s1 = {};
#pragma unroll
    for (int ds = 0; ds < 4; ++ds) {
      v16bf k0 = *(const v16bf*)(Kh + kvbase + (size_t)(kc + l15) * D_ + ds * 32 + lh * 16);
      v16bf k1 = *(const v16bf*)(Kh + kvbase + (size_t)(kc + 16 + l15) * D_ + ds * 32 + lh * 16);
      s0 = wmma_bf16f32(qf[ds], k0, s0);
      s1 = wmma_bf16f32(qf[ds], k1, s1);
    }
    float pr0[8], pr1[8];
#pragma unroll
    for (int r = 0; r < 8; ++r) {
      float v0 = s0[r] * scale, v1 = s1[r] * scale;
      float mx = fmaxf(v0, v1);
#pragma unroll
      for (int off = 8; off >= 1; off >>= 1) mx = fmaxf(mx, __shfl_xor(mx, off, 32));
      float mnew = fmaxf(mi[r], mx);
      float p0 = __expf(v0 - mnew), p1 = __expf(v1 - mnew);
      float rs = p0 + p1;
#pragma unroll
      for (int off = 8; off >= 1; off >>= 1) rs += __shfl_xor(rs, off, 32);
      float f = __expf(mi[r] - mnew);
      li[r] = li[r] * f + rs;
      mi[r] = mnew;
#pragma unroll
      for (int tt = 0; tt < 8; ++tt) o[tt][r] *= f;
      pr0[r] = p0; pr1[r] = p1;
    }
    // C-layout -> A-layout transpose of P via LDS (wave-private, no barrier)
#pragma unroll
    for (int r = 0; r < 8; ++r) {
      int prow = r + lh * 8;
      pbuf[wid][prow * 32 + l15]      = (__bf16)pr0[r];
      pbuf[wid][prow * 32 + 16 + l15] = (__bf16)pr1[r];
    }
    v16bf pa = make_afrag(&pbuf[wid][l15 * 32 + lh * 16]);
#pragma unroll
    for (int tt = 0; tt < 8; ++tt) {
      v16bf vb = *(const v16bf*)(Vt + kvbase + (size_t)(tt * 16 + l15) * L_ + kc + lh * 16);
      o[tt] = wmma_bf16f32(pa, vb, o[tt]);
    }
  }
#pragma unroll
  for (int r = 0; r < 8; ++r) {
    float inv = 1.0f / li[r];
    int gq = q0 + r + lh * 8;
    size_t yo = ((size_t)b * L_ + gq) * K2_ + hd * D_;
#pragma unroll
    for (int tt = 0; tt < 8; ++tt) Y[yo + tt * 16 + l15] = (__bf16)(o[tt][r] * inv);
  }
}

// ---------------- launch ----------------
extern "C" void kernel_launch(void* const* d_in, const int* in_sizes, int n_in,
                              void* d_out, int out_size, void* d_ws, size_t ws_size,
                              hipStream_t stream) {
  const float* x       = (const float*)d_in[0];
  const float* vec     = (const float*)d_in[1];
  const float* pe      = (const float*)d_in[2];
  const float* mod_w   = (const float*)d_in[3];
  const float* mod_b   = (const float*)d_in[4];
  const float* lin1_w  = (const float*)d_in[5];
  const float* lin1_b  = (const float*)d_in[6];
  const float* lin2_w  = (const float*)d_in[7];
  const float* lin2_b  = (const float*)d_in[8];
  const float* q_scale = (const float*)d_in[9];
  const float* k_scale = (const float*)d_in[10];
  float* out = (float*)d_out;
  (void)in_sizes; (void)n_in; (void)out_size; (void)ws_size;

  char* base = (char*)d_ws;
  size_t off = 0;
  auto alloc = [&](size_t bytes) -> void* {
    void* p = base + off;
    off = (off + bytes + 255) & ~(size_t)255;
    return p;
  };
  float*  modbuf = (float*)alloc((size_t)B_ * 3 * H_ * sizeof(float));
  __bf16* xmod   = (__bf16*)alloc((size_t)M_ * H_ * 2);
  __bf16* w1     = (__bf16*)alloc((size_t)N1_ * H_ * 2);
  __bf16* w2     = (__bf16*)alloc((size_t)H_ * K2_ * 2);
  float*  hbuf   = (float*)alloc((size_t)M_ * N1_ * sizeof(float));
  __bf16* qh     = (__bf16*)alloc((size_t)B_ * NH_ * L_ * D_ * 2);
  __bf16* kh     = (__bf16*)alloc((size_t)B_ * NH_ * L_ * D_ * 2);
  __bf16* vt     = (__bf16*)alloc((size_t)B_ * NH_ * L_ * D_ * 2);
  __bf16* ybuf   = (__bf16*)alloc((size_t)M_ * K2_ * 2);

  cvt_bf16<<<(unsigned)(((size_t)N1_ * H_) / 256), 256, 0, stream>>>(lin1_w, w1, (size_t)N1_ * H_);
  cvt_bf16<<<(unsigned)(((size_t)H_ * K2_) / 256), 256, 0, stream>>>(lin2_w, w2, (size_t)H_ * K2_);
  mod_kernel<<<(B_ * 3 * H_) / 256, 256, 0, stream>>>(vec, mod_w, mod_b, modbuf);
  ln_mod<<<M_, 256, 0, stream>>>(x, modbuf, xmod);
  gemm_bf16<false><<<dim3(M_ / 128, N1_ / 128), 256, 0, stream>>>(
      xmod, w1, lin1_b, hbuf, nullptr, nullptr, M_, N1_, H_);
  qkv_prep<<<M_, 256, 0, stream>>>(hbuf, pe, q_scale, k_scale, qh, kh, vt);
  gelu_kernel<<<(unsigned)(((size_t)M_ * MLP_) / 256), 256, 0, stream>>>(hbuf, ybuf);
  attn_kernel<<<dim3(B_ * NH_, L_ / 64), 128, 0, stream>>>(qh, kh, vt, ybuf);
  gemm_bf16<true><<<dim3(M_ / 128, H_ / 128), 256, 0, stream>>>(
      ybuf, w2, lin2_b, out, x, modbuf, M_, H_, K2_);
}
